// BnnMnistModel_48653389529541
// MI455X (gfx1250) — matrix operand order repaired and checked by reading, not compiled, not run on hardware
//
#include <hip/hip_runtime.h>
#include <cstdint>

typedef __attribute__((ext_vector_type(16))) _Float16 v16h;
typedef __attribute__((ext_vector_type(8)))  float    v8f;
typedef __attribute__((ext_vector_type(4)))  float    v4f;

#define BN_EPS 1e-3f
#define USE_ASYNC_LDS 1

#define WAVES 8
#define TM 16                              // rows per wave (one M-tile)
#define ROWS_PER_BLOCK (WAVES * TM)        // 128
#define KPAD 800                           // 784 padded to 25 k-steps of 32
#define NP1 64                             // layer-1 N (50) padded to 4 n-tiles

// Low 32 bits of a generic LDS pointer are the wave-relative LDS byte offset
// (see ISA section 10.2).
__device__ __forceinline__ uint32_t lds_off(const void* p) {
  return (uint32_t)(uintptr_t)p;
}

__global__ __launch_bounds__(256) void bnn_fused(
    const float* __restrict__ X,
    const float* __restrict__ W1, const float* __restrict__ W2, const float* __restrict__ W3,
    const float* __restrict__ g1, const float* __restrict__ b1,
    const float* __restrict__ m1, const float* __restrict__ v1,
    const float* __restrict__ g2, const float* __restrict__ b2,
    const float* __restrict__ m2, const float* __restrict__ v2,
    float* __restrict__ out, int Bn)
{
  // sign(W) in exact f16 {+1,-1}, stored n-major so B-fragments are contiguous K runs.
  __shared__ __align__(128) _Float16 sW1T[NP1][KPAD];   // 100 KB
  __shared__ __align__(128) _Float16 sW2T[64][64];      //   8 KB
  __shared__ __align__(128) _Float16 sW3T[16][64];      //   2 KB
  __shared__ __align__(128) unsigned char sScr[WAVES][4096];  // 32 KB per-wave scratch

  const int tid  = threadIdx.x;
  const int lane = tid & 31;
  const int wave = tid >> 5;

  // ---------------- Phase 0: binarize weights into LDS ----------------
  for (int idx = tid; idx < NP1 * KPAD; idx += 256) {
    int k = idx >> 6;          // n fast -> coalesced W1 reads (L2-resident)
    int n = idx & 63;
    _Float16 val = (_Float16)0.f;
    if (k < 784 && n < 50)
      val = (W1[k * 50 + n] >= 0.f) ? (_Float16)1.f : (_Float16)-1.f;
    sW1T[n][k] = val;
  }
  for (int idx = tid; idx < 64 * 64; idx += 256) {
    int k = idx >> 6, n = idx & 63;
    _Float16 val = (_Float16)0.f;
    if (k < 50 && n < 50)
      val = (W2[k * 50 + n] >= 0.f) ? (_Float16)1.f : (_Float16)-1.f;
    sW2T[n][k] = val;
  }
  for (int idx = tid; idx < 64 * 16; idx += 256) {
    int k = idx >> 4, n = idx & 15;
    _Float16 val = (_Float16)0.f;
    if (k < 50 && n < 10)
      val = (W3[k * 10 + n] >= 0.f) ? (_Float16)1.f : (_Float16)-1.f;
    sW3T[n][k] = val;
  }
  __syncthreads();

  const int row0 = blockIdx.x * ROWS_PER_BLOCK + wave * TM;
  float* xb0 = (float*)&sScr[wave][0];      // [16][32] chunk buffer A
  float* xb1 = (float*)&sScr[wave][2048];   // [16][32] chunk buffer B

  const int m  = lane & 15;   // A-frag row / C-frag column / B-frag column
  const int kh = lane >> 4;   // K-half selector (A/B) and M-half selector (C)

  // Stage one 16-row x 32-K f32 chunk of X into LDS (2 KB = 4 x 512B bursts).
  auto stage = [&](float* xbuf, int kk) {
#if USE_ASYNC_LDS
    asm volatile("s_wait_dscnt 0" ::: "memory");  // WAR vs prior reads of this buffer
    uint32_t lb = lds_off(xbuf);
    #pragma unroll
    for (int i = 0; i < 4; ++i) {
      int flat = i * 512 + lane * 16;   // byte offset within the 2 KB tile
      int row  = flat >> 7;             // 128 B per row
      int cb   = flat & 127;
      int r = row0 + row; if (r > Bn - 1) r = Bn - 1;
      const char* gp = (const char*)(X + (size_t)r * 784 + kk * 32) + cb;
      asm volatile("global_load_async_to_lds_b128 %0, %1, off"
                   :: "v"(lb + (uint32_t)flat), "v"((uint64_t)(uintptr_t)gp)
                   : "memory");
    }
#else
    #pragma unroll
    for (int i = 0; i < 4; ++i) {
      int flat = i * 512 + lane * 16;
      int row = flat >> 7, cb = flat & 127;
      int r = row0 + row; if (r > Bn - 1) r = Bn - 1;
      *(v4f*)((char*)xbuf + flat) =
          *(const v4f*)((const char*)(X + (size_t)r * 784 + kk * 32) + cb);
    }
#endif
  };

  v8f acc[4] = {};   // layer-1 accumulators, 4 n-tiles of 16

  // One K-step (K=32) of layer 1: A split into hi/lo f16 (exact vs ±1 weights).
  auto compute = [&](const float* xbuf, int kbase) {
    const v4f* xp = (const v4f*)(xbuf + m * 32 + kh * 8);
    v4f x0 = xp[0], x1 = xp[1], x2 = xp[4], x3 = xp[5];
    float xs[16];
    #pragma unroll
    for (int j = 0; j < 4; ++j) { xs[j]=x0[j]; xs[4+j]=x1[j]; xs[8+j]=x2[j]; xs[12+j]=x3[j]; }
    v16h ahi, alo;
    #pragma unroll
    for (int e = 0; e < 16; ++e) {
      _Float16 h = (_Float16)xs[e];
      ahi[e] = h;
      alo[e] = (_Float16)(xs[e] - (float)h);
    }
    #pragma unroll
    for (int t = 0; t < 4; ++t) {
      v16h b; const _Float16* bp = &sW1T[t * 16 + m][kbase + kh * 8];
      for (int e = 0; e < 8; ++e) { b[e] = bp[e]; b[8 + e] = bp[16 + e]; }
      acc[t] = __builtin_amdgcn_wmma_f32_16x16x32_f16(false, ahi, false, b, (short)0, acc[t], false, false);
      acc[t] = __builtin_amdgcn_wmma_f32_16x16x32_f16(false, alo, false, b, (short)0, acc[t], false, false);
    }
  };

  // ---------------- Layer 1: K-steps 0..23 double-buffered through LDS ----------------
  stage(xb0, 0);
  for (int kk = 0; kk < 24; ++kk) {
    float* cur = (kk & 1) ? xb1 : xb0;
    if (kk < 23) {
      stage((kk & 1) ? xb0 : xb1, kk + 1);
#if USE_ASYNC_LDS
      asm volatile("s_wait_asynccnt 4" ::: "memory");  // previous chunk's 4 DMAs done
#endif
    } else {
#if USE_ASYNC_LDS
      asm volatile("s_wait_asynccnt 0" ::: "memory");
#endif
    }
    compute(cur, kk * 32);
  }

  // K-step 24 (k = 768..799): first 8-half chunk valid, second is zero padding.
  {
    int r = row0 + m; if (r > Bn - 1) r = Bn - 1;
    const v4f* gp = (const v4f*)(X + (size_t)r * 784 + 768 + kh * 8);
    v4f x0 = gp[0], x1 = gp[1];
    v16h ahi, alo;
    #pragma unroll
    for (int e = 0; e < 16; ++e) { ahi[e] = (_Float16)0.f; alo[e] = (_Float16)0.f; }
    float xs[8];
    #pragma unroll
    for (int j = 0; j < 4; ++j) { xs[j] = x0[j]; xs[4 + j] = x1[j]; }
    #pragma unroll
    for (int e = 0; e < 8; ++e) {
      _Float16 h = (_Float16)xs[e];
      ahi[e] = h;
      alo[e] = (_Float16)(xs[e] - (float)h);
    }
    #pragma unroll
    for (int t = 0; t < 4; ++t) {
      v16h b; const _Float16* bp = &sW1T[t * 16 + m][768 + kh * 8];
      for (int e = 0; e < 8; ++e) { b[e] = bp[e]; b[8 + e] = bp[16 + e]; }
      acc[t] = __builtin_amdgcn_wmma_f32_16x16x32_f16(false, ahi, false, b, (short)0, acc[t], false, false);
      acc[t] = __builtin_amdgcn_wmma_f32_16x16x32_f16(false, alo, false, b, (short)0, acc[t], false, false);
    }
  }

  _Float16* hbuf  = (_Float16*)&sScr[wave][0];      // [16][64] sign activations
  float*    lgbuf = (float*)&sScr[wave][2048];      // [16][16] logits

  // ---------------- Epilogue 1: relu -> BN1 -> sign, C-layout -> A-layout via LDS ----
  #pragma unroll
  for (int t = 0; t < 4; ++t) {
    int col = t * 16 + m;
    float s = 0.f, o = -1.f;
    if (col < 50) {
      float sv = g1[col] * rsqrtf(v1[col] + BN_EPS);
      s = sv; o = b1[col] - m1[col] * sv;
    }
    #pragma unroll
    for (int e = 0; e < 8; ++e) {
      int mr = kh * 8 + e;
      float h = acc[t][e]; h = h > 0.f ? h : 0.f;
      _Float16 val = (_Float16)0.f;
      if (col < 50) val = (s * h + o >= 0.f) ? (_Float16)1.f : (_Float16)-1.f;
      hbuf[mr * 64 + col] = val;   // cols >= 50 padded with 0
    }
  }

  // ---------------- Layer 2: [16x64]±1 @ [64x64]±1 (exact in f16) ----------------
  v8f acc2[4] = {};
  #pragma unroll
  for (int kk = 0; kk < 2; ++kk) {
    const _Float16* ap = hbuf + m * 64 + kk * 32 + kh * 8;
    v16h a;
    #pragma unroll
    for (int e = 0; e < 8; ++e) { a[e] = ap[e]; a[8 + e] = ap[16 + e]; }
    #pragma unroll
    for (int t = 0; t < 4; ++t) {
      v16h b; const _Float16* bp = &sW2T[t * 16 + m][kk * 32 + kh * 8];
      for (int e = 0; e < 8; ++e) { b[e] = bp[e]; b[8 + e] = bp[16 + e]; }
      acc2[t] = __builtin_amdgcn_wmma_f32_16x16x32_f16(false, a, false, b, (short)0, acc2[t], false, false);
    }
  }

  // ---------------- Epilogue 2: relu -> BN2 -> sign ----------------
  #pragma unroll
  for (int t = 0; t < 4; ++t) {
    int col = t * 16 + m;
    float s = 0.f, o = -1.f;
    if (col < 50) {
      float sv = g2[col] * rsqrtf(v2[col] + BN_EPS);
      s = sv; o = b2[col] - m2[col] * sv;
    }
    #pragma unroll
    for (int e = 0; e < 8; ++e) {
      int mr = kh * 8 + e;
      float h = acc2[t][e]; h = h > 0.f ? h : 0.f;
      _Float16 val = (_Float16)0.f;
      if (col < 50) val = (s * h + o >= 0.f) ? (_Float16)1.f : (_Float16)-1.f;
      hbuf[mr * 64 + col] = val;
    }
  }

  // ---------------- Layer 3: logits = sign(h) @ sign(W3), one 16-wide n-tile -------
  v8f acc3 = {};
  #pragma unroll
  for (int kk = 0; kk < 2; ++kk) {
    const _Float16* ap = hbuf + m * 64 + kk * 32 + kh * 8;
    v16h a;
    #pragma unroll
    for (int e = 0; e < 8; ++e) { a[e] = ap[e]; a[8 + e] = ap[16 + e]; }
    v16h b; const _Float16* bp = &sW3T[m][kk * 32 + kh * 8];
    for (int e = 0; e < 8; ++e) { b[e] = bp[e]; b[8 + e] = bp[16 + e]; }
    acc3 = __builtin_amdgcn_wmma_f32_16x16x32_f16(false, a, false, b, (short)0, acc3, false, false);
  }
  #pragma unroll
  for (int e = 0; e < 8; ++e)
    lgbuf[(kh * 8 + e) * 16 + m] = acc3[e];

  // ---------------- Softmax: one row per lane (lanes 0..15) ----------------
  if (lane < 16) {
    int r = row0 + lane;
    if (r < Bn) {
      float l[10], mx = -3.4e38f;
      #pragma unroll
      for (int j = 0; j < 10; ++j) { l[j] = lgbuf[lane * 16 + j]; mx = fmaxf(mx, l[j]); }
      float sum = 0.f;
      #pragma unroll
      for (int j = 0; j < 10; ++j) { l[j] = expf(l[j] - mx); sum += l[j]; }
      float inv = 1.f / sum;
      #pragma unroll
      for (int j = 0; j < 10; ++j) out[(size_t)r * 10 + j] = l[j] * inv;
    }
  }
}

extern "C" void kernel_launch(void* const* d_in, const int* in_sizes, int n_in,
                              void* d_out, int out_size, void* d_ws, size_t ws_size,
                              hipStream_t stream) {
  (void)n_in; (void)out_size; (void)d_ws; (void)ws_size;
  const float* X  = (const float*)d_in[0];
  const float* W1 = (const float*)d_in[1];
  const float* W2 = (const float*)d_in[2];
  const float* W3 = (const float*)d_in[3];
  const float* g1 = (const float*)d_in[4];
  const float* b1 = (const float*)d_in[5];
  const float* m1 = (const float*)d_in[6];
  const float* v1 = (const float*)d_in[7];
  const float* g2 = (const float*)d_in[8];
  const float* b2 = (const float*)d_in[9];
  const float* m2 = (const float*)d_in[10];
  const float* v2 = (const float*)d_in[11];

  int Bn   = in_sizes[0] / 784;
  int grid = (Bn + ROWS_PER_BLOCK - 1) / ROWS_PER_BLOCK;
  bnn_fused<<<grid, 256, 0, stream>>>(X, W1, W2, W3,
                                      g1, b1, m1, v1, g2, b2, m2, v2,
                                      (float*)d_out, Bn);
}
